// TransformerBlock1_85830626443903
// MI455X (gfx1250) — compile-verified, matrix operand order checked by reading, not run on hardware
//
#include <hip/hip_runtime.h>

#define BN   8
#define NP   2048
#define KNN  20
#define XR   67      // x row stride (3 xyz + 64 feat)
#define EPS_F 1e-5f

typedef __attribute__((ext_vector_type(16))) _Float16 v16h;
typedef __attribute__((ext_vector_type(8)))  float    v8f;
typedef __attribute__((ext_vector_type(4)))  unsigned int u32x4;
typedef __attribute__((ext_vector_type(8)))  int      i32x8;
typedef __attribute__((ext_vector_type(4)))  int      i32x4;

union V16H { v16h v; float4 q[2]; };

// Load one WMMA f16 operand in the documented 16-bit 16x32 layout:
// per lane: 8 halves at [off], 8 halves at [off+16].
__device__ __forceinline__ v16h ld_ab(const _Float16* p, int off) {
  V16H u;
  u.q[0] = *(const float4*)(p + off);
  u.q[1] = *(const float4*)(p + off + 16);
  return u.v;
}

__device__ __forceinline__ float wsum(float v) {
#pragma unroll
  for (int m = 16; m >= 1; m >>= 1) v += __shfl_xor(v, m, 32);
  return v;
}

// ---------------- K0: convert + pad weights to f16 in workspace ----------------
__global__ __launch_bounds__(256) void prep_weights(
    const float* __restrict__ W1, const float* __restrict__ W2,
    const float* __restrict__ Wq, const float* __restrict__ Wv,
    const float* __restrict__ Wk,
    _Float16* __restrict__ W1p, _Float16* __restrict__ W2p,
    _Float16* __restrict__ Wqp, _Float16* __restrict__ Wvp,
    _Float16* __restrict__ Wkp) {
  int t = blockIdx.x * 256 + threadIdx.x;
  if (t < 64 * 160)  { int m = t / 160, k = t - m * 160; W1p[t] = (_Float16)(k < 144 ? W1[m * 144 + k] : 0.f); }
  if (t < 64 * 64)   W2p[t] = (_Float16)W2[t];
  if (t < 256 * 64)  Wqp[t] = (_Float16)Wq[t];
  if (t < 16 * 160)  { int m = t / 160, k = t - m * 160; Wvp[t] = (_Float16)(k < 144 ? Wv[m * 144 + k] : 0.f); }
  if (t < 32 * 160)  { int m = t / 160, k = t - m * 160; Wkp[t] = (_Float16)(k < 144 ? Wk[m * 144 + k] : 0.f); }
}

// ---------------- K1: brute-force KNN (top-20), 1 wave per query ----------------
// xyz staged into LDS by the Tensor Data Mover: 2-D tile (3 x 2048 dwords),
// tensor row stride 67 dwords, packed into LDS -> pxyz[n*3+c].
__global__ __launch_bounds__(128) void knn_kernel(const float* __restrict__ x,
                                                  int* __restrict__ idxp,
                                                  float* __restrict__ out_xyz) {
  __shared__ float pxyz[NP * 3];
  __shared__ unsigned long long mbuf[4][32 * KNN];
  const int tid = threadIdx.x;
  const int b  = blockIdx.x >> 9;          // 512 blocks per batch
  const int n0 = (blockIdx.x & 511) << 2;  // 4 queries per block
  const float* xg = x + (size_t)b * NP * XR;

  if (tid < 32) {  // wave 0 issues the TDM descriptor
    unsigned long long ga = (unsigned long long)(uintptr_t)xg;
    unsigned int ldsoff = (unsigned int)(uintptr_t)&pxyz[0];  // low 32 bits = LDS offset
    u32x4 g0 = { 1u,                                   // count=1 (valid user D#)
                 ldsoff,                               // lds_addr
                 (unsigned int)(ga & 0xFFFFFFFFu),     // global_addr[31:0]
                 (unsigned int)((ga >> 32) & 0x01FFFFFFu) | 0x80000000u }; // ga[56:32] | type=2
    i32x8 g1 = { (int)(2u << 16),          // data_size=2 (4B), no multicast/pad/iter
                 (int)((XR & 0xFFFFu) << 16),          // tensor_dim0 = 67 (low16 -> [31:16])
                 (int)(((unsigned)NP & 0xFFFFu) << 16),// dim0 hi=0 | tensor_dim1=2048 lo16
                 (int)(3u << 16),                      // tensor_dim1 hi=0 | tile_dim0=3
                 (int)NP,                              // tile_dim1=2048 | tile_dim2=0
                 (int)XR,                              // tensor_dim0_stride = 67 (low 32)
                 (int)(((unsigned)(XR * NP) & 0xFFFFu) << 16), // d0s hi=0 | d1_stride lo16
                 (int)((unsigned)(XR * NP) >> 16) };   // tensor_dim1_stride hi
    i32x4 gz4 = { 0, 0, 0, 0 };
    i32x8 gz8 = { 0, 0, 0, 0, 0, 0, 0, 0 };
    __builtin_amdgcn_tensor_load_to_lds(g0, g1, gz4, gz4, gz8, 0);
    __builtin_amdgcn_s_wait_tensorcnt((short)0);
  }
  __syncthreads();

  if (tid < 12) {  // xyz passthrough output
    int p = tid / 3, c = tid - p * 3;
    out_xyz[((size_t)b * NP + n0 + p) * 3 + c] = pxyz[(n0 + p) * 3 + c];
  }

  const int wv = tid >> 5, ln = tid & 31;
  const int q = n0 + wv;
  const float qx = pxyz[q * 3 + 0], qy = pxyz[q * 3 + 1], qz = pxyz[q * 3 + 2];

  float kd[KNN]; int ki[KNN];
#pragma unroll
  for (int j = 0; j < KNN; ++j) { kd[j] = 3.4e38f; ki[j] = 0; }

  for (int t = 0; t < NP / 32; ++t) {
    int ci = t * 32 + ln;
    float dx = pxyz[ci * 3 + 0] - qx;
    float dy = pxyz[ci * 3 + 1] - qy;
    float dz = pxyz[ci * 3 + 2] - qz;
    float d = dx * dx + dy * dy + dz * dz;
#pragma unroll
    for (int j = KNN - 1; j >= 0; --j) {
      if (d < kd[j]) {
        if (j < KNN - 1) { kd[j + 1] = kd[j]; ki[j + 1] = ki[j]; }
        kd[j] = d; ki[j] = ci;
      }
    }
  }
#pragma unroll
  for (int j = 0; j < KNN; ++j)
    mbuf[wv][ln * KNN + j] =
        ((unsigned long long)__float_as_uint(kd[j]) << 32) | (unsigned)ki[j];
  __syncthreads();

  int cnt = 0;
  for (int r = 0; r < KNN; ++r) {
    unsigned long long head = (cnt < KNN) ? mbuf[wv][ln * KNN + cnt] : ~0ULL;
    unsigned long long best = head;
#pragma unroll
    for (int m = 16; m >= 1; m >>= 1) {
      unsigned long long o = __shfl_xor(best, m, 32);
      if (o < best) best = o;
    }
    if (head == best && cnt < KNN) ++cnt;
    if (ln == 0) idxp[((size_t)b * NP + q) * KNN + r] = (int)(best & 0xFFFFFFFFULL);
  }
}

// ---------------- K2: fused group-MLP + attention, 16 points per block ----------------
__global__ __launch_bounds__(256) void fused_kernel(
    const float* __restrict__ x, const float* __restrict__ Wh,
    const float* __restrict__ g1, const float* __restrict__ b1,
    const float* __restrict__ g2, const float* __restrict__ b2,
    const float* __restrict__ lnqw, const float* __restrict__ lnqb,
    const float* __restrict__ lnvw_g, const float* __restrict__ lnvb_g,
    const float* __restrict__ lnow, const float* __restrict__ lnob,
    const int* __restrict__ idxp,
    const _Float16* __restrict__ W1p, const _Float16* __restrict__ W2p,
    const _Float16* __restrict__ Wqp, const _Float16* __restrict__ Wvp,
    const _Float16* __restrict__ Wkp,
    float* __restrict__ outp) {
  __shared__ _Float16 kkall[KNN * 16 * 32];  // softmax logits / weights
  __shared__ _Float16 vall[KNN * 16 * 16];   // LN'd v values
  __shared__ float    fq[16 * 64];           // maxpool accumulator
  __shared__ _Float16 fqh[16 * 64];
  __shared__ int      idxs[16 * KNN];
  union UAB {
    struct { _Float16 F[16 * 160]; _Float16 h1[16 * 64]; } a;  // chunk-loop phase
    struct { float q[16 * 256]; float ob[16 * 128]; } bp;      // post-loop phase
  };
  __shared__ UAB u;

  const int tid = threadIdx.x;
  const int wv = tid >> 5, ln = tid & 31, lm = ln & 15, lh = ln >> 4;
  const int b  = blockIdx.x >> 7;          // 128 tiles per batch
  const int n0 = (blockIdx.x & 127) << 4;  // 16 points per block
  const float* xg = x + (size_t)b * NP * XR;

  __builtin_prefetch(W1p); __builtin_prefetch(Wqp); __builtin_prefetch(Wkp);

  // per-wave BN/LN parameter preload + loop-invariant WMMA A-operands in registers
  float s1[8], t1[8], s2[8], t2[8], vw[8], vb[8];
  v16h a1[5];   // layer1 / Wv / Wk operand (5 K-chunks of 32)
  v16h a2[2];   // layer2 operand (2 K-chunks)
  const float binv = rsqrtf(1.f + EPS_F);
  if (wv < 4) {
#pragma unroll
    for (int r = 0; r < 8; ++r) {
      int o = wv * 16 + r + 8 * lh;
      s1[r] = g1[o] * binv; t1[r] = b1[o];
      s2[r] = g2[o] * binv; t2[r] = b2[o];
    }
#pragma unroll
    for (int kc = 0; kc < 5; ++kc)
      a1[kc] = ld_ab(W1p + (wv * 16 + lm) * 160, kc * 32 + lh * 8);
#pragma unroll
    for (int kc = 0; kc < 2; ++kc)
      a2[kc] = ld_ab(W2p + (wv * 16 + lm) * 64, kc * 32 + lh * 8);
  } else if (wv == 4) {
#pragma unroll
    for (int r = 0; r < 8; ++r) { int c = r + 8 * lh; vw[r] = lnvw_g[c]; vb[r] = lnvb_g[c]; }
#pragma unroll
    for (int kc = 0; kc < 5; ++kc)
      a1[kc] = ld_ab(Wvp + lm * 160, kc * 32 + lh * 8);
  } else if (wv < 7) {
#pragma unroll
    for (int kc = 0; kc < 5; ++kc)
      a1[kc] = ld_ab(Wkp + ((wv - 5) * 16 + lm) * 160, kc * 32 + lh * 8);
  }

  for (int i = tid; i < 16 * KNN; i += 256)
    idxs[i] = idxp[((size_t)b * NP + n0 + i / KNN) * KNN + (i % KNN)];
  for (int i = tid; i < 16 * 64; i += 256) fq[i] = -3.4e38f;
  __syncthreads();

  const int fn = tid >> 4;   // point column this thread builds
  const int c0 = tid & 15;   // channel phase
  const float* crow = xg + (size_t)(n0 + fn) * XR;

  for (int k = 0; k < KNN; ++k) {
    // --- build F column tile (16 cols x 160 padded channels, f16) ---
    {
      int nbr = idxs[fn * KNN + k];
      const float* nrow = xg + (size_t)nbr * XR;
      float rx = nrow[0] - crow[0], ry = nrow[1] - crow[1], rz = nrow[2] - crow[2];
#pragma unroll
      for (int ci = 0; ci < 10; ++ci) {
        int c = c0 + ci * 16;
        float val;
        if (c < 64)        val = nrow[3 + c] - crow[3 + c];
        else if (c < 128)  val = crow[3 + (c - 64)];
        else if (c < 144) { int j = c - 128;
          val = Wh[j * 3] * rx + Wh[j * 3 + 1] * ry + Wh[j * 3 + 2] * rz; }
        else               val = 0.f;
        u.a.F[fn * 160 + c] = (_Float16)val;
      }
    }
    __syncthreads();

    if (wv < 7) {  // waves 0-6 consume F with WMMA; preload all B chunks first
      v16h Bs[5];
#pragma unroll
      for (int kc = 0; kc < 5; ++kc)
        Bs[kc] = ld_ab(u.a.F + lm * 160, kc * 32 + lh * 8);
      v8f acc = {};
#pragma unroll
      for (int kc = 0; kc < 5; ++kc)
        acc = __builtin_amdgcn_wmma_f32_16x16x32_f16(false, a1[kc], false, Bs[kc], (short)0, acc, false, false);

      if (wv < 4) {             // layer1 epilogue: BN + relu -> h1
#pragma unroll
        for (int r = 0; r < 8; ++r) {
          float hv = fmaxf(acc[r] * s1[r] + t1[r], 0.f);
          u.a.h1[lm * 64 + wv * 16 + r + 8 * lh] = (_Float16)hv;
        }
      } else if (wv == 4) {     // v epilogue: LN over 16 channels
        float s = 0.f, s2v = 0.f;
#pragma unroll
        for (int r = 0; r < 8; ++r) { s += acc[r]; s2v += acc[r] * acc[r]; }
        s   += __shfl_xor(s, 16, 32);
        s2v += __shfl_xor(s2v, 16, 32);
        float mean = s * 0.0625f;
        float rs = rsqrtf(s2v * 0.0625f - mean * mean + EPS_F);
#pragma unroll
        for (int r = 0; r < 8; ++r) {
          float nv = (acc[r] - mean) * rs * vw[r] + vb[r];
          vall[(k * 16 + lm) * 16 + r + 8 * lh] = (_Float16)nv;
        }
      } else {                  // kk epilogue: raw logits
        int dt = wv - 5;
#pragma unroll
        for (int r = 0; r < 8; ++r)
          kkall[(k * 16 + lm) * 32 + dt * 16 + r + 8 * lh] = (_Float16)acc[r];
      }
    }
    __syncthreads();

    if (wv < 4) {               // layer2: 64x64 . 64x16 (2 WMMA) -> running max
      v16h Bs[2];
#pragma unroll
      for (int kc = 0; kc < 2; ++kc)
        Bs[kc] = ld_ab(u.a.h1 + lm * 64, kc * 32 + lh * 8);
      v8f acc = {};
#pragma unroll
      for (int kc = 0; kc < 2; ++kc)
        acc = __builtin_amdgcn_wmma_f32_16x16x32_f16(false, a2[kc], false, Bs[kc], (short)0, acc, false, false);
#pragma unroll
      for (int r = 0; r < 8; ++r) {
        int o = wv * 16 + r + 8 * lh;
        float hv = fmaxf(acc[r] * s2[r] + t2[r], 0.f);
        float* pf = &fq[lm * 64 + o];
        *pf = fmaxf(*pf, hv);
      }
    }
    __syncthreads();
  }

  for (int i = tid; i < 16 * 64; i += 256) fqh[i] = (_Float16)fq[i];
  __syncthreads();

  // q = W_q (256x64) . fq (64x16); each wave handles 2 row tiles
#pragma unroll
  for (int mi = 0; mi < 2; ++mi) {
    int mt = wv + mi * 8;
    v16h Aq[2], Bs[2];
#pragma unroll
    for (int kc = 0; kc < 2; ++kc) {
      Aq[kc] = ld_ab(Wqp + (mt * 16 + lm) * 64, kc * 32 + lh * 8);
      Bs[kc] = ld_ab(fqh + lm * 64, kc * 32 + lh * 8);
    }
    v8f acc = {};
#pragma unroll
    for (int kc = 0; kc < 2; ++kc)
      acc = __builtin_amdgcn_wmma_f32_16x16x32_f16(false, Aq[kc], false, Bs[kc], (short)0, acc, false, false);
#pragma unroll
    for (int r = 0; r < 8; ++r)
      u.bp.q[lm * 256 + mt * 16 + r + 8 * lh] = acc[r];
  }
  __syncthreads();

  // LN + relu on q (over 256 channels per point)
  for (int nn = wv; nn < 16; nn += 8) {
    float xs[8], s = 0.f, s2v = 0.f;
#pragma unroll
    for (int j = 0; j < 8; ++j) {
      xs[j] = u.bp.q[nn * 256 + ln + 32 * j];
      s += xs[j]; s2v += xs[j] * xs[j];
    }
    s = wsum(s); s2v = wsum(s2v);
    float mean = s * (1.f / 256.f);
    float rs = rsqrtf(s2v * (1.f / 256.f) - mean * mean + EPS_F);
#pragma unroll
    for (int j = 0; j < 8; ++j) {
      int c = ln + 32 * j;
      float v = (xs[j] - mean) * rs * lnqw[c] + lnqb[c];
      u.bp.q[nn * 256 + c] = fmaxf(v, 0.f);
    }
  }
  __syncthreads();

  // softmax over the 20 neighbors per (n, d)
  for (int task = tid; task < 16 * 32; task += 256) {
    int nn = task >> 5, d = task & 31;
    float mx = -3.4e38f;
#pragma unroll
    for (int k = 0; k < KNN; ++k) mx = fmaxf(mx, (float)kkall[(k * 16 + nn) * 32 + d]);
    float ev[KNN], sum = 0.f;
#pragma unroll
    for (int k = 0; k < KNN; ++k) {
      ev[k] = __expf((float)kkall[(k * 16 + nn) * 32 + d] - mx);
      sum += ev[k];
    }
    float inv = 1.f / sum;
#pragma unroll
    for (int k = 0; k < KNN; ++k) kkall[(k * 16 + nn) * 32 + d] = (_Float16)(ev[k] * inv);
  }
  __syncthreads();

  // out[n,h,:] = sum_k (sum_d q[n,h,d]*kk[n,k,d]) * v[n,k,:]   (kv never materialized)
  if (tid < 128) {
    int nn = tid >> 3, h = tid & 7;
    float qv[32];
#pragma unroll
    for (int d = 0; d < 32; ++d) qv[d] = u.bp.q[nn * 256 + h * 32 + d];
    float a_[KNN];
#pragma unroll
    for (int k = 0; k < KNN; ++k) {
      float s = 0.f;
#pragma unroll
      for (int d = 0; d < 32; ++d) s += qv[d] * (float)kkall[(k * 16 + nn) * 32 + d];
      a_[k] = s;
    }
#pragma unroll
    for (int vv = 0; vv < 16; ++vv) {
      float s = 0.f;
#pragma unroll
      for (int k = 0; k < KNN; ++k) s += a_[k] * (float)vall[(k * 16 + nn) * 16 + vv];
      u.bp.ob[nn * 128 + h * 16 + vv] = s;
    }
  }
  __syncthreads();

  // final LN over 128 channels + transposed store (B,128,N)
  for (int nn = wv; nn < 16; nn += 8) {
    float xs[4], s = 0.f, s2v = 0.f;
#pragma unroll
    for (int j = 0; j < 4; ++j) {
      xs[j] = u.bp.ob[nn * 128 + ln + 32 * j];
      s += xs[j]; s2v += xs[j] * xs[j];
    }
    s = wsum(s); s2v = wsum(s2v);
    float mean = s * (1.f / 128.f);
    float rs = rsqrtf(s2v * (1.f / 128.f) - mean * mean + EPS_F);
#pragma unroll
    for (int j = 0; j < 4; ++j) {
      int c = ln + 32 * j;
      float v = (xs[j] - mean) * rs * lnow[c] + lnob[c];
      outp[((size_t)b * 128 + c) * NP + (n0 + nn)] = v;
    }
  }
}

extern "C" void kernel_launch(void* const* d_in, const int* in_sizes, int n_in,
                              void* d_out, int out_size, void* d_ws, size_t ws_size,
                              hipStream_t stream) {
  const float* x    = (const float*)d_in[0];
  const float* Wh   = (const float*)d_in[1];
  const float* W1   = (const float*)d_in[2];
  const float* g1   = (const float*)d_in[3];
  const float* b1   = (const float*)d_in[4];
  const float* W2   = (const float*)d_in[5];
  const float* g2   = (const float*)d_in[6];
  const float* b2   = (const float*)d_in[7];
  const float* Wq   = (const float*)d_in[8];
  const float* lnqw = (const float*)d_in[9];
  const float* lnqb = (const float*)d_in[10];
  const float* Wv   = (const float*)d_in[11];
  const float* lnvw = (const float*)d_in[12];
  const float* lnvb = (const float*)d_in[13];
  const float* Wk   = (const float*)d_in[14];
  const float* lnow = (const float*)d_in[15];
  const float* lnob = (const float*)d_in[16];

  char* ws = (char*)d_ws;
  int*      idxp = (int*)ws;                         // 8*2048*20*4 = 1,310,720 B
  _Float16* W1p  = (_Float16*)(ws + 1310720);        // 64*160*2
  _Float16* W2p  = (_Float16*)(ws + 1331200);        // 64*64*2
  _Float16* Wqp  = (_Float16*)(ws + 1339392);        // 256*64*2
  _Float16* Wvp  = (_Float16*)(ws + 1372160);        // 16*160*2
  _Float16* Wkp  = (_Float16*)(ws + 1377280);        // 32*160*2

  float* out_xyz = (float*)d_out;
  float* outp    = (float*)d_out + (size_t)BN * NP * 3;

  prep_weights<<<64, 256, 0, stream>>>(W1, W2, Wq, Wv, Wk, W1p, W2p, Wqp, Wvp, Wkp);
  knn_kernel<<<BN * (NP / 4), 128, 0, stream>>>(x, idxp, out_xyz);
  fused_kernel<<<BN * (NP / 16), 256, 0, stream>>>(
      x, Wh, g1, b1, g2, b2, lnqw, lnqb, lnvw, lnvb, lnow, lnob,
      idxp, W1p, W2p, Wqp, Wvp, Wkp, outp);
}
